// AdaptiveGraphConvolution_59365037965633
// MI455X (gfx1250) — matrix operand order repaired and checked by reading, not compiled, not run on hardware
//
#include <hip/hip_runtime.h>
#include <hip/hip_bf16.h>
#include <stdint.h>

// ---------------- CDNA5 WMMA types ----------------
typedef __attribute__((ext_vector_type(16))) __bf16 v16bf;
typedef __attribute__((ext_vector_type(8)))  float  v8f;

union Frag {
    v16bf  v;
    __bf16 h[16];
    uint4  q[2];
};

__device__ __forceinline__ v8f zero_v8f() {
    v8f z;
#pragma unroll
    for (int i = 0; i < 8; ++i) z[i] = 0.0f;
    return z;
}

__device__ __forceinline__ v8f wmma_bf16(const v16bf a, const v16bf b, const v8f c) {
    // D = A(16x32 bf16) * B(32x16 bf16) + C(16x16 f32)
    return __builtin_amdgcn_wmma_f32_16x16x32_bf16(false, a, false, b, (short)0, c, false, false);
}

// CDNA5 async global->LDS copy (ASYNCcnt tracked, in-order completion per wave)
__device__ __forceinline__ void async_ld_b32(unsigned lds_addr, unsigned gvoff, const float* sbase) {
    asm volatile("global_load_async_to_lds_b32 %0, %1, %2"
                 :
                 : "v"(lds_addr), "v"(gvoff), "s"(sbase)
                 : "memory");
}
#define WAIT_ASYNC(n) asm volatile("s_wait_asynccnt " #n ::: "memory")

// ---------------- problem constants ----------------
#define N_   64
#define C_   64
#define T_   300
#define V_   25
#define IC_  16
#define NS_  3
#define O_   64
#define TV_  (T_ * V_)        // 7500
#define TCH1 30               // t's per workgroup in kernel 1
#define NCH1 (T_ / TCH1)      // 10 chunks
#define TCH3 10               // t's per workgroup in kernel 3
#define NCH3 (T_ / TCH3)      // 30 chunks

// logits_part layout: [chunk(10)][i(3)][n(64)][u(32)][w(32)] f32
// attnT layout:       [i*64+n][w(32)][u(32)] bf16  (transposed, zero-padded)

// =====================================================================
// Kernel 1: fused a/b 1x1 conv (96x64 GEMM) + attention logits (K=4800)
// =====================================================================
__global__ __launch_bounds__(256) void k_conv_logits(
    const float* __restrict__ x,
    const float* __restrict__ a_w, const float* __restrict__ a_b,
    const float* __restrict__ b_w, const float* __restrict__ b_b,
    float* __restrict__ logits_part)
{
    __shared__ __align__(16) __bf16 wlds[96][64];        // fused [a0 b0 a1 b1 a2 b2] weights
    __shared__ float                 biasl[96];
    __shared__ __align__(16) float   xf1[2][3200];       // async staging, 2 t's f32 (dbl buf)
    __shared__ __align__(16) __bf16 xsT[2][32][64];      // x tile, [tpair][u(pad32)][c]
    __shared__ __align__(16) __bf16 abT[2][32][96];      // conv out, [tpair][col][row]

    const int n     = blockIdx.y;
    const int chunk = blockIdx.x;
    const int tid   = threadIdx.x;
    const int wv    = tid >> 5;
    const int lane  = tid & 31;

    const float* xn = x + (size_t)n * C_ * TV_;

    // ---- async prefetch of the first t-pair into buffer 0 ----
    {
        const int t0 = chunk * TCH1;
        const int nIter = (wv < 4) ? 13 : 12;            // 3200 = 12.5 * 256, wave-uniform
        for (int j = 0; j < nIter; ++j) {
            int idx = tid + j * 256;
            int tp  = idx / 1600;
            int rem = idx - tp * 1600;
            int c   = rem / 25;
            int u   = rem - c * 25;
            unsigned lds = (unsigned)(uintptr_t)&xf1[0][idx];
            unsigned go  = (unsigned)((c * TV_ + (t0 + tp) * V_ + u) * 4);
            async_ld_b32(lds, go, xn);
        }
    }

    // ---- load fused weights / biases into LDS (bf16); zero xsT pad rows once ----
    for (int idx = tid; idx < 96 * 64; idx += 256) {
        int r = idx >> 6, c = idx & 63;
        int i = r >> 5, rr = r & 31;
        float w = (rr < 16) ? a_w[(i * 16 + rr) * 64 + c]
                            : b_w[(i * 16 + (rr - 16)) * 64 + c];
        wlds[r][c] = (__bf16)w;
    }
    if (tid < 96) {
        int i = tid >> 5, rr = tid & 31;
        biasl[tid] = (rr < 16) ? a_b[i * 16 + rr] : b_b[i * 16 + rr - 16];
    }
    for (int idx = tid; idx < 2 * 7 * 64; idx += 256) {  // u >= 25 pad rows (never rewritten)
        int tp  = idx / 448;
        int rem = idx - tp * 448;
        int u   = 25 + rem / 64;
        int c   = rem & 63;
        xsT[tp][u][c] = (__bf16)0.0f;
    }
    __syncthreads();

    // ---- per-wave conv tiles: tau = wv*3 + j ; tau -> (tp, mt, nn) ----
    Frag wA[3][2];
    const int tauBase = wv * 3;
#pragma unroll
    for (int j = 0; j < 3; ++j) {
        int tau = tauBase + j;
        int mt  = (tau % 12) >> 1;
        int r   = mt * 16 + (lane & 15);
        int kb  = (lane < 16) ? 0 : 16;         // A-matrix K split
#pragma unroll
        for (int kt = 0; kt < 2; ++kt) {
            wA[j][kt].q[0] = *(const uint4*)&wlds[r][kt * 32 + kb];
            wA[j][kt].q[1] = *(const uint4*)&wlds[r][kt * 32 + kb + 8];
        }
    }

    // ---- logits accumulators: 12 tiles = 3 subsets x (2 u-tiles x 2 w-tiles) ----
    const int ntile = (wv < 4) ? 2 : 1;
    v8f lacc[2];
    lacc[0] = zero_v8f();
    lacc[1] = zero_v8f();

    for (int it = 0; it < TCH1 / 2; ++it) {
        const int cur = it & 1;

        // issue async prefetch of next t-pair into the other buffer, then wait
        // until the current buffer's loads (issued one step earlier) are done.
        if (it + 1 < TCH1 / 2) {
            const int t0n = chunk * TCH1 + (it + 1) * 2;
            const int nIter = (wv < 4) ? 13 : 12;
            for (int j = 0; j < nIter; ++j) {
                int idx = tid + j * 256;
                int tp  = idx / 1600;
                int rem = idx - tp * 1600;
                int c   = rem / 25;
                int u   = rem - c * 25;
                unsigned lds = (unsigned)(uintptr_t)&xf1[cur ^ 1][idx];
                unsigned go  = (unsigned)((c * TV_ + (t0n + tp) * V_ + u) * 4);
                async_ld_b32(lds, go, xn);
            }
            if (wv < 4) { WAIT_ASYNC(13); } else { WAIT_ASYNC(12); }
        } else {
            WAIT_ASYNC(0);
        }
        __syncthreads();

        // convert staged f32 -> transposed bf16 tile xsT[tp][u][c]
        for (int idx = tid; idx < 3200; idx += 256) {
            int tp  = idx / 1600;
            int rem = idx - tp * 1600;
            int c   = rem / 25;
            int u   = rem - c * 25;
            xsT[tp][u][c] = (__bf16)xf1[cur][idx];
        }
        __syncthreads();

        // ---- conv: D = W(96x64) * x_t(64x32), 3 tiles per wave ----
#pragma unroll
        for (int j = 0; j < 3; ++j) {
            int tau = tauBase + j;
            int tp  = tau / 12;
            int rem = tau % 12;
            int mt  = rem >> 1;
            int nn  = rem & 1;
            int ucol = nn * 16 + (lane & 15);
            int kb   = (lane < 16) ? 0 : 16;    // B-matrix K split

            Frag xB0, xB1;
            xB0.q[0] = *(const uint4*)&xsT[tp][ucol][kb];
            xB0.q[1] = *(const uint4*)&xsT[tp][ucol][kb + 8];
            xB1.q[0] = *(const uint4*)&xsT[tp][ucol][32 + kb];
            xB1.q[1] = *(const uint4*)&xsT[tp][ucol][32 + kb + 8];

            v8f acc = zero_v8f();
            acc = wmma_bf16(wA[j][0].v, xB0.v, acc);
            acc = wmma_bf16(wA[j][1].v, xB1.v, acc);

            // store transposed + bias: abT[tp][col][row]
            int rbase = mt * 16 + ((lane < 16) ? 0 : 8);
#pragma unroll
            for (int e = 0; e < 8; ++e) {
                int r = rbase + e;
                abT[tp][ucol][r] = (__bf16)(acc[e] + biasl[r]);
            }
        }
        __syncthreads();

        // ---- logits: logit[u,w] += sum_{tp,ic} a[tp,ic,u]*b[tp,ic,w] (K=32) ----
        for (int s = 0; s < ntile; ++s) {
            int tau = wv + s * 8;
            int i   = tau >> 2;
            int mu  = (tau & 3) >> 1;
            int nw  = tau & 1;
            int u    = mu * 16 + (lane & 15);
            int wcol = nw * 16 + (lane & 15);

            // A-frag: rows=u, K=(tp,ic)  -> abT[tp][u][i*32 + ic]
            int icb = (lane < 16) ? 0 : 8;
            Frag aA;
            aA.q[0] = *(const uint4*)&abT[0][u][i * 32 + icb];
            aA.q[1] = *(const uint4*)&abT[1][u][i * 32 + icb];

            // B-frag: cols=w, K=(tp,ic)  -> abT[tp][w][i*32 + 16 + ic]
            int icB = (lane < 16) ? 0 : 8;
            Frag bB;
            bB.q[0] = *(const uint4*)&abT[0][wcol][i * 32 + 16 + icB];
            bB.q[1] = *(const uint4*)&abT[1][wcol][i * 32 + 16 + icB];

            lacc[s] = wmma_bf16(aA.v, bB.v, lacc[s]);
        }
        __syncthreads();
    }

    // ---- write per-chunk partial logits (deterministic; no atomics) ----
    for (int s = 0; s < ntile; ++s) {
        int tau = wv + s * 8;
        int i   = tau >> 2;
        int mu  = (tau & 3) >> 1;
        int nw  = tau & 1;
        int wcol  = nw * 16 + (lane & 15);
        int rbase = mu * 16 + ((lane < 16) ? 0 : 8);
        float* lp = logits_part + ((((size_t)chunk * NS_ + i) * N_ + n) * 32) * 32;
#pragma unroll
        for (int e = 0; e < 8; ++e) {
            lp[(size_t)(rbase + e) * 32 + wcol] = lacc[s][e];
        }
    }
}

// =====================================================================
// Kernel 2: reduce partials, softmax over u (axis=-2), + A + graph_attn,
//           write transposed zero-padded bf16 attn (B-fragment friendly)
// =====================================================================
__global__ __launch_bounds__(32) void k_softmax(
    const float* __restrict__ logits_part,
    const float* __restrict__ A, const float* __restrict__ gattn,
    __bf16* __restrict__ attnT)
{
    const int b = blockIdx.x;          // i*64 + n
    const int i = b / N_;
    const int n = b - i * N_;
    const int w = threadIdx.x;         // column
    __bf16* dst = attnT + ((size_t)b * 32 + w) * 32;

    if (w < V_) {
        float col[V_];
        float mx = -1e30f;
#pragma unroll
        for (int u = 0; u < V_; ++u) {
            float s = 0.0f;
#pragma unroll
            for (int ch = 0; ch < NCH1; ++ch) {
                s += logits_part[((((size_t)ch * NS_ + i) * N_ + n) * 32 + u) * 32 + w];
            }
            s *= (1.0f / (float)(IC_ * T_));
            col[u] = s;
            mx = fmaxf(mx, s);
        }
        float sum = 0.0f;
#pragma unroll
        for (int u = 0; u < V_; ++u) { col[u] = __expf(col[u] - mx); sum += col[u]; }
        float inv = 1.0f / sum;
#pragma unroll
        for (int u = 0; u < V_; ++u) {
            float v = col[u] * inv
                    + A[(i * V_ + u) * V_ + w]
                    + gattn[(i * V_ + u) * V_ + w];
            dst[u] = (__bf16)v;
        }
#pragma unroll
        for (int u = V_; u < 32; ++u) dst[u] = (__bf16)0.0f;   // zero K-pad
    } else {
#pragma unroll
        for (int u = 0; u < 32; ++u) dst[u] = (__bf16)0.0f;    // pad columns
    }
}

// =====================================================================
// Kernel 3: fused  z = x_t @ attn_i  ->  g_conv  -> sum_i -> BN+res+ReLU
// =====================================================================
__global__ __launch_bounds__(256) void k_main(
    const float* __restrict__ x, const __bf16* __restrict__ attnT,
    const float* __restrict__ g_w, const float* __restrict__ g_b,
    const float* __restrict__ bn_gamma, const float* __restrict__ bn_beta,
    const float* __restrict__ bn_mean,  const float* __restrict__ bn_var,
    float* __restrict__ out)
{
    __shared__ __align__(16) float   xf3[2][64][25];   // async-staged x_t f32 (dbl buf; residual)
    __shared__ __align__(16) __bf16 xs3[64][32];       // x_t [c][u(pad32)] bf16
    __shared__ __align__(16) __bf16 zT[32][64];        // z transposed [v][c] bf16
    __shared__ float scL[64], shL[64];                 // folded BN params

    const int n     = blockIdx.y;
    const int chunk = blockIdx.x;
    const int tid   = threadIdx.x;
    const int wv    = tid >> 5;
    const int lane  = tid & 31;
    const int mo    = wv >> 1;     // output row tile (also z row tile)
    const int nv    = wv & 1;      // output col tile

    const float* xn = x   + (size_t)n * C_ * TV_;
    float*       on = out + (size_t)n * O_ * TV_;

    // ---- async prefetch of first t into buffer 0 ----
    {
        const int t0 = chunk * TCH3;
        const int nIter = (wv < 2) ? 7 : 6;             // 1600 = 6.25 * 256, wave-uniform
        for (int j = 0; j < nIter; ++j) {
            int idx = tid + j * 256;
            int c = idx / 25, u = idx - c * 25;
            unsigned lds = (unsigned)(uintptr_t)&xf3[0][c][u];
            unsigned go  = (unsigned)((c * TV_ + t0 * V_ + u) * 4);
            async_ld_b32(lds, go, xn);
        }
    }

    if (tid < 64) {
        float gbs = g_b[tid] + g_b[64 + tid] + g_b[128 + tid];   // sum_i g_b[i][o]
        float sc  = bn_gamma[tid] * rsqrtf(bn_var[tid] + 1e-5f);
        scL[tid]  = sc;
        shL[tid]  = (gbs - bn_mean[tid]) * sc + bn_beta[tid];
    }
    for (int idx = tid; idx < 64 * 7; idx += 256) {      // xs3 pad cols, written once
        int c = idx / 7, u = 25 + idx % 7;
        xs3[c][u] = (__bf16)0.0f;
    }

    // ---- preload per-wave register fragments ----
    Frag attnB[3];         // B: K=u(32), cols = v-tile
    Frag gA[3][2];         // A: rows = o-tile, K = c (2 x K32)
    {
        const int vcol = nv * 16 + (lane & 15);
        const int ub   = (lane < 16) ? 0 : 16;     // B-matrix K split
#pragma unroll
        for (int i = 0; i < NS_; ++i) {
            const __bf16* ap = attnT + (((size_t)(i * N_ + n)) * 32 + vcol) * 32;
            attnB[i].q[0] = *(const uint4*)(ap + ub);
            attnB[i].q[1] = *(const uint4*)(ap + ub + 8);
        }
        const int orow = mo * 16 + (lane & 15);
        const int cb   = (lane < 16) ? 0 : 16;     // A-matrix K split
#pragma unroll
        for (int i = 0; i < NS_; ++i) {
#pragma unroll
            for (int kt = 0; kt < 2; ++kt) {
                const float* gp = g_w + ((size_t)i * 64 + orow) * 64 + kt * 32 + cb;
#pragma unroll
                for (int e = 0; e < 8; ++e) {
                    gA[i][kt].h[e]     = (__bf16)gp[e];
                    gA[i][kt].h[8 + e] = (__bf16)gp[8 + e];
                }
            }
        }
    }

    for (int tt = 0; tt < TCH3; ++tt) {
        const int t   = chunk * TCH3 + tt;
        const int cur = tt & 1;

        // prefetch next t into the other buffer, wait for current buffer
        if (tt + 1 < TCH3) {
            const int tnext = t + 1;
            const int nIter = (wv < 2) ? 7 : 6;
            for (int j = 0; j < nIter; ++j) {
                int idx = tid + j * 256;
                int c = idx / 25, u = idx - c * 25;
                unsigned lds = (unsigned)(uintptr_t)&xf3[cur ^ 1][c][u];
                unsigned go  = (unsigned)((c * TV_ + tnext * V_ + u) * 4);
                async_ld_b32(lds, go, xn);
            }
            if (wv < 2) { WAIT_ASYNC(7); } else { WAIT_ASYNC(6); }
        } else {
            WAIT_ASYNC(0);
        }
        __syncthreads();

        // convert staged f32 -> bf16 xs3 (pad cols already zero)
        for (int idx = tid; idx < 64 * 25; idx += 256) {
            int c = idx / 25, u = idx - c * 25;
            xs3[c][u] = (__bf16)xf3[cur][c][u];
        }
        __syncthreads();

        // x A-frag for this wave's row tile (K = u, one K32 covers padded V)
        Frag xA;
        {
            int crow = mo * 16 + (lane & 15);
            int ub   = (lane < 16) ? 0 : 16;
            xA.q[0] = *(const uint4*)&xs3[crow][ub];
            xA.q[1] = *(const uint4*)&xs3[crow][ub + 8];
        }

        v8f h = zero_v8f();
#pragma unroll
        for (int i = 0; i < NS_; ++i) {
            // z tile = x_t @ attn_i  (single WMMA, K=32)
            v8f zacc = zero_v8f();
            zacc = wmma_bf16(xA.v, attnB[i].v, zacc);
            __syncthreads();                       // prior zT readers done
            {
                int vcol = nv * 16 + (lane & 15);
                int rb   = mo * 16 + ((lane < 16) ? 0 : 8);
#pragma unroll
                for (int e = 0; e < 8; ++e) zT[vcol][rb + e] = (__bf16)zacc[e];
            }
            __syncthreads();                       // zT ready

            // hidden += g_w[i] @ z   (K=64 -> 2 WMMAs)
            int vcol = nv * 16 + (lane & 15);
            int cb2  = (lane < 16) ? 0 : 16;
            Frag zB0, zB1;
            zB0.q[0] = *(const uint4*)&zT[vcol][cb2];
            zB0.q[1] = *(const uint4*)&zT[vcol][cb2 + 8];
            zB1.q[0] = *(const uint4*)&zT[vcol][32 + cb2];
            zB1.q[1] = *(const uint4*)&zT[vcol][32 + cb2 + 8];
            h = wmma_bf16(gA[i][0].v, zB0.v, h);
            h = wmma_bf16(gA[i][1].v, zB1.v, h);
            __syncthreads();                       // zB reads done before next overwrite
        }

        // BN (+folded g biases) + residual + ReLU, store f32
        {
            int vcol = nv * 16 + (lane & 15);
            if (vcol < V_) {
                int rb = mo * 16 + ((lane < 16) ? 0 : 8);
#pragma unroll
                for (int e = 0; e < 8; ++e) {
                    int o = rb + e;
                    float val = h[e] * scL[o] + shL[o] + xf3[cur][o][vcol];
                    on[o * TV_ + t * V_ + vcol] = fmaxf(val, 0.0f);
                }
            }
        }
        __syncthreads();                           // xf3/xs3 reads done before re-stage
    }
}

// =====================================================================
extern "C" void kernel_launch(void* const* d_in, const int* in_sizes, int n_in,
                              void* d_out, int out_size, void* d_ws, size_t ws_size,
                              hipStream_t stream) {
    (void)in_sizes; (void)n_in; (void)out_size; (void)ws_size;

    const float* x        = (const float*)d_in[0];
    const float* A        = (const float*)d_in[1];
    const float* gattn    = (const float*)d_in[2];
    const float* g_w      = (const float*)d_in[3];
    const float* g_b      = (const float*)d_in[4];
    const float* a_w      = (const float*)d_in[5];
    const float* a_b      = (const float*)d_in[6];
    const float* b_w      = (const float*)d_in[7];
    const float* b_b      = (const float*)d_in[8];
    const float* bn_gamma = (const float*)d_in[9];
    const float* bn_beta  = (const float*)d_in[10];
    const float* bn_mean  = (const float*)d_in[11];
    const float* bn_var   = (const float*)d_in[12];
    float* out = (float*)d_out;

    // workspace layout
    float*  logits_part = (float*)d_ws;                                  // 10*3*64*32*32 f32 = 7.5 MiB
    __bf16* attnT = (__bf16*)((char*)d_ws +
                              (size_t)NCH1 * NS_ * N_ * 32 * 32 * sizeof(float));

    k_conv_logits<<<dim3(NCH1, N_), 256, 0, stream>>>(x, a_w, a_b, b_w, b_b, logits_part);
    k_softmax<<<dim3(NS_ * N_), 32, 0, stream>>>(logits_part, A, gattn, attnT);
    k_main<<<dim3(NCH3, N_), 256, 0, stream>>>(x, attnT, g_w, g_b,
                                               bn_gamma, bn_beta, bn_mean, bn_var, out);
}
